// TransformGroup_56968446214730
// MI455X (gfx1250) — compile-verified, matrix operand order checked by reading, not run on hardware
//
#include <hip/hip_runtime.h>

// ---------------------------------------------------------------- constants
#define N_T     64
#define D_MODEL 1024
#define RANK    8
#define BATCH   4096
#define KTOT    (N_T * RANK)      // 512 stacked low-rank features
#define BTILE   32                // batch rows per workgroup
#define XLD     1032              // LDS row pitch for x tile (bf16), 4-bank rotate
#define GLD     68                // LDS row pitch for gate tile (f32)
#define ZLD     520               // LDS row pitch for Z' tile (bf16)

typedef __attribute__((ext_vector_type(16))) __bf16 bf16x16;
typedef __attribute__((ext_vector_type(8)))  float  f32x8;

// ---------------------------------------------------------------- helpers
__device__ __forceinline__ f32x8 wmma_bf16(bf16x16 a, bf16x16 b, f32x8 c) {
    // v_wmma_f32_16x16x32_bf16: D = A(16x32) * B(32x16) + C(16x16 f32)
    return __builtin_amdgcn_wmma_f32_16x16x32_bf16(
        /*neg_a=*/false, a, /*neg_b=*/false, b,
        /*c_mod=*/(short)0, c, /*reuse_a=*/false, /*reuse_b=*/false);
}

// Load a 16x32 bf16 fragment from a row-major bf16 matrix (pitch `ld` elems).
// ISA 7.12.2: lane L(0..15) = row (row0+L), K = k0+0..7 and k0+16..23;
// lane L+16 = same row, K shifted by +8.  Same addressing serves A (rows of X)
// and B when the B source is stored feature-major (rows = B^T rows).
__device__ __forceinline__ bf16x16 ldfrag(const __bf16* __restrict__ base,
                                          int row0, int ld, int k0) {
    const int lane = threadIdx.x & 31;
    const int r    = lane & 15;
    const int h    = lane >> 4;
    const __bf16* p = base + (size_t)(row0 + r) * ld + k0 + h * 8;
    union { bf16x16 v; uint4 q[2]; } u;
    u.q[0] = *(const uint4*)(p);
    u.q[1] = *(const uint4*)(p + 16);
    return u.v;
}

// ---------------------------------------------------------------- pre-pass
__global__ void k_cvt_bf16(const float* __restrict__ in,
                           __bf16* __restrict__ out, int n4) {
    int i = blockIdx.x * blockDim.x + threadIdx.x;
    if (i >= n4) return;
    float4 v = ((const float4*)in)[i];
    union { __bf16 h[4]; uint2 u; } o;
    o.h[0] = (__bf16)v.x; o.h[1] = (__bf16)v.y;
    o.h[2] = (__bf16)v.z; o.h[3] = (__bf16)v.w;
    ((uint2*)out)[i] = o.u;
}

// U (N,D,R) f32 -> Ustack^T (D x KTOT) bf16: ubt[d*512 + n*8 + r] = U[n,d,r]
__global__ void k_cvt_ut(const float* __restrict__ U, __bf16* __restrict__ ubt) {
    int idx = blockIdx.x * blockDim.x + threadIdx.x;   // 65536 = 64*1024
    int n = idx >> 10;
    int d = idx & 1023;
    const float* p = U + (size_t)n * (D_MODEL * RANK) + (size_t)d * RANK;
    union { __bf16 h[8]; uint4 u; } o;
#pragma unroll
    for (int r = 0; r < 8; ++r) o.h[r] = (__bf16)p[r];
    *(uint4*)(ubt + (size_t)d * KTOT + n * RANK) = o.u;
}

// ---------------------------------------------------------------- fused main
__global__ __launch_bounds__(256)
void k_fused(const __bf16* __restrict__ xb,    // (BATCH x D_MODEL) bf16
             const __bf16* __restrict__ vb,    // (KTOT  x D_MODEL) bf16  (Vstack)
             const __bf16* __restrict__ ubt,   // (D_MODEL x KTOT)  bf16  (Ustack^T)
             const __bf16* __restrict__ encb,  // (N_T   x D_MODEL) bf16
             const float*  __restrict__ bias,  // (N_T)
             float* __restrict__ out,          // (BATCH x D_MODEL)
             float* __restrict__ gate_out) {   // (BATCH x N_T)
    extern __shared__ char smem[];
    __bf16* xs = (__bf16*)smem;                                       // BTILE x XLD
    float*  gs = (float*) (smem + BTILE * XLD * 2);                   // BTILE x GLD
    __bf16* zs = (__bf16*)(smem + BTILE * XLD * 2 + BTILE * GLD * 4); // BTILE x ZLD

    const int tid  = threadIdx.x;
    const int w    = tid >> 5;          // wave id 0..7
    const int lane = tid & 31;
    const int r16  = lane & 15;
    const int h    = lane >> 4;
    const int b0   = blockIdx.x * BTILE;

    // ---- stage x tile into LDS via CDNA5 async global->LDS path (ASYNCcnt)
    for (int c = tid; c < BTILE * (D_MODEL / 8); c += 256) {
        int row = c >> 7;               // 128 16B-chunks per row
        int col = (c & 127) * 8;
        unsigned lds_off = (unsigned)(uintptr_t)(xs + row * XLD + col);
        const __bf16* g = xb + (size_t)(b0 + row) * D_MODEL + col;
        asm volatile("global_load_async_to_lds_b128 %0, %1, off"
                     :: "v"(lds_off), "v"(g) : "memory");
    }
    asm volatile("s_wait_asynccnt 0x0" ::: "memory");
    __syncthreads();

    // ---- phase 1a: pre_acts -> gate (8 tiles of 16x16, one per wave)
    {
        const int mt = w & 1;
        const int n0 = (w >> 1) * 16;
        f32x8 acc = {};
#pragma unroll 4
        for (int k0 = 0; k0 < D_MODEL; k0 += 32) {
            bf16x16 a  = ldfrag(xs,   mt * 16, XLD,     k0);
            bf16x16 bf = ldfrag(encb, n0,      D_MODEL, k0);
            acc = wmma_bf16(a, bf, acc);
        }
        const int   col  = n0 + r16;
        const float bcol = bias[col];
#pragma unroll
        for (int i = 0; i < 8; ++i) {
            int   m   = mt * 16 + h * 8 + i;
            float pre = acc[i] - bcol;                 // x@enc^T - bias
            float g   = pre > 0.f ? pre : 0.f;         // jumprelu(thr=0)
            gs[m * GLD + col] = g;
            gate_out[(size_t)(b0 + m) * N_T + col] = g;
        }
    }
    __syncthreads();

    // ---- phase 1b: Z = X @ Vstack^T, 2x2 register blocking, gate -> zs
    for (int blk = w; blk < 16; blk += 8) {            // 16 blocks of 2mt x 2nt
        const int n0 = blk * 32;
        f32x8 a00 = {}, a01 = {}, a10 = {}, a11 = {};
#pragma unroll 4
        for (int k0 = 0; k0 < D_MODEL; k0 += 32) {
            bf16x16 fa0 = ldfrag(xs, 0,  XLD, k0);
            bf16x16 fa1 = ldfrag(xs, 16, XLD, k0);
            bf16x16 fb0 = ldfrag(vb, n0,      D_MODEL, k0);
            bf16x16 fb1 = ldfrag(vb, n0 + 16, D_MODEL, k0);
            a00 = wmma_bf16(fa0, fb0, a00);
            a01 = wmma_bf16(fa0, fb1, a01);
            a10 = wmma_bf16(fa1, fb0, a10);
            a11 = wmma_bf16(fa1, fb1, a11);
        }
#pragma unroll
        for (int i = 0; i < 8; ++i) {
            int m0 = h * 8 + i;                        // rows 0..15
            int m1 = 16 + m0;                          // rows 16..31
            int c0 = n0 + r16;                         // cols n0..n0+15
            int c1 = c0 + 16;
            zs[m0 * ZLD + c0] = (__bf16)(a00[i] * gs[m0 * GLD + (c0 >> 3)]);
            zs[m0 * ZLD + c1] = (__bf16)(a01[i] * gs[m0 * GLD + (c1 >> 3)]);
            zs[m1 * ZLD + c0] = (__bf16)(a10[i] * gs[m1 * GLD + (c0 >> 3)]);
            zs[m1 * ZLD + c1] = (__bf16)(a11[i] * gs[m1 * GLD + (c1 >> 3)]);
        }
    }
    __syncthreads();

    // ---- phase 2: Out = Z' @ Ustack, 2x2 register blocking
    for (int blk = w; blk < 32; blk += 8) {            // 32 blocks of 2mt x 2dt
        const int d0 = blk * 32;
        f32x8 a00 = {}, a01 = {}, a10 = {}, a11 = {};
#pragma unroll 4
        for (int k0 = 0; k0 < KTOT; k0 += 32) {
            bf16x16 fa0 = ldfrag(zs, 0,  ZLD, k0);
            bf16x16 fa1 = ldfrag(zs, 16, ZLD, k0);
            bf16x16 fb0 = ldfrag(ubt, d0,      KTOT, k0);
            bf16x16 fb1 = ldfrag(ubt, d0 + 16, KTOT, k0);
            a00 = wmma_bf16(fa0, fb0, a00);
            a01 = wmma_bf16(fa0, fb1, a01);
            a10 = wmma_bf16(fa1, fb0, a10);
            a11 = wmma_bf16(fa1, fb1, a11);
        }
#pragma unroll
        for (int i = 0; i < 8; ++i) {
            int m0 = h * 8 + i;
            int m1 = 16 + m0;
            int c0 = d0 + r16;
            int c1 = c0 + 16;
            out[(size_t)(b0 + m0) * D_MODEL + c0] = a00[i];
            out[(size_t)(b0 + m0) * D_MODEL + c1] = a01[i];
            out[(size_t)(b0 + m1) * D_MODEL + c0] = a10[i];
            out[(size_t)(b0 + m1) * D_MODEL + c1] = a11[i];
        }
    }
}

// ---------------------------------------------------------------- frobenius
__global__ void k_fro(const float* __restrict__ U, const float* __restrict__ V,
                      float* __restrict__ fro) {
    __shared__ float red[256];
    const int n = blockIdx.x;
    const float* u = U + (size_t)n * (D_MODEL * RANK);
    const float* v = V + (size_t)n * (D_MODEL * RANK);
    float su = 0.f, sv = 0.f;
    for (int i = threadIdx.x; i < D_MODEL * RANK; i += 256) {
        float a = u[i]; su += a * a;
        float b = v[i]; sv += b * b;
    }
    red[threadIdx.x] = su; __syncthreads();
    for (int s = 128; s > 0; s >>= 1) {
        if (threadIdx.x < s) red[threadIdx.x] += red[threadIdx.x + s];
        __syncthreads();
    }
    float tsu = red[0]; __syncthreads();
    red[threadIdx.x] = sv; __syncthreads();
    for (int s = 128; s > 0; s >>= 1) {
        if (threadIdx.x < s) red[threadIdx.x] += red[threadIdx.x + s];
        __syncthreads();
    }
    if (threadIdx.x == 0)
        fro[n] = sqrtf(tsu) * sqrtf(red[0]) / sqrtf((float)(D_MODEL * RANK));
}

// ---------------------------------------------------------------- launch
extern "C" void kernel_launch(void* const* d_in, const int* in_sizes, int n_in,
                              void* d_out, int out_size, void* d_ws, size_t ws_size,
                              hipStream_t stream) {
    const float* x    = (const float*)d_in[0];   // (4096,1024)
    const float* V    = (const float*)d_in[1];   // (64,8,1024)
    const float* U    = (const float*)d_in[2];   // (64,1024,8)
    const float* enc  = (const float*)d_in[3];   // (64,1024)
    const float* bias = (const float*)d_in[4];   // (64,)

    float* out      = (float*)d_out;                       // (4096,1024)
    float* gate_out = out + (size_t)BATCH * D_MODEL;       // (4096,64)
    float* fro      = gate_out + (size_t)BATCH * N_T;      // (64,)

    char* ws = (char*)d_ws;
    __bf16* xb   = (__bf16*)(ws);                          // 8 MB
    __bf16* vb   = (__bf16*)(ws + (8u << 20));             // 1 MB
    __bf16* ubt  = (__bf16*)(ws + (9u << 20));             // 1 MB
    __bf16* encb = (__bf16*)(ws + (10u << 20));            // 128 KB

    // bf16 pre-pass
    {
        int n4 = BATCH * D_MODEL / 4;
        k_cvt_bf16<<<(n4 + 255) / 256, 256, 0, stream>>>(x, xb, n4);
        n4 = KTOT * D_MODEL / 4;
        k_cvt_bf16<<<(n4 + 255) / 256, 256, 0, stream>>>(V, vb, n4);
        n4 = N_T * D_MODEL / 4;
        k_cvt_bf16<<<(n4 + 255) / 256, 256, 0, stream>>>(enc, encb, n4);
        k_cvt_ut<<<(N_T * D_MODEL) / 256, 256, 0, stream>>>(U, ubt);
    }

    // fused gate + low-rank mixture
    size_t smem = (size_t)BTILE * XLD * 2 + (size_t)BTILE * GLD * 4
                + (size_t)BTILE * ZLD * 2;                 // ~105.5 KB
    k_fused<<<BATCH / BTILE, 256, smem, stream>>>(xb, vb, ubt, encb, bias,
                                                  out, gate_out);

    // frobenius norms (independent)
    k_fro<<<N_T, 256, 0, stream>>>(U, V, fro);
}
